// AttnBlockpp_62182536512142
// MI455X (gfx1250) — matrix operand order repaired and checked by reading, not compiled, not run on hardware
//
#include <hip/hip_runtime.h>
#include <hip/hip_bf16.h>

#define C_DIM 256
#define N_DIM 4096
#define B_DIM 4

typedef __attribute__((ext_vector_type(16))) _Float16 v16h;
typedef __attribute__((ext_vector_type(8)))  _Float16 v8h;
typedef __attribute__((ext_vector_type(8)))  float    v8f;
typedef __attribute__((ext_vector_type(4)))  unsigned int u32x4;
typedef __attribute__((ext_vector_type(8)))  int      i32x8;

__device__ __forceinline__ v8f wmma_f16(v16h a, v16h b, v8f c) {
  // D = A(16x32 f16) * B(32x16 f16) + C(16x16 f32)
  return __builtin_amdgcn_wmma_f32_16x16x32_f16(false, a, false, b, (short)0, c,
                                                false, false);
}

__device__ __forceinline__ v16h pack16(v8h lo, v8h hi) {
  // straight concat: keeps the two b128 loads in adjacent VGPR quads
  return __builtin_shufflevector(lo, hi, 0, 1, 2, 3, 4, 5, 6, 7, 8, 9, 10, 11,
                                 12, 13, 14, 15);
}

// ---------------------------------------------------------------------------
// TDM: DMA a 2-D tile (row-major, 4-byte elements) from global into LDS with
// optional per-row padding, per CDNA5 D# descriptor layout (group0 + group1).
// Issue from ONE wave per workgroup; completion via s_wait_tensorcnt.
// ---------------------------------------------------------------------------
__device__ __forceinline__ void tdm_load_2d(unsigned long long gaddr,
                                            unsigned int lds_off,
                                            unsigned int tensor_d0,   // dwords/row
                                            unsigned int tensor_d1,   // rows
                                            unsigned int stride0,     // dwords
                                            unsigned int tile_d0,     // dwords
                                            unsigned int tile_d1,     // rows
                                            unsigned int pad_interval_code,
                                            unsigned int pad_amount_code,
                                            unsigned int pad_enable) {
  u32x4 g0;
  g0[0] = 1u;                                   // count=1 (valid user D#)
  g0[1] = lds_off;                              // lds_addr [63:32]
  g0[2] = (unsigned int)gaddr;                  // global_addr [95:64]
  g0[3] = (unsigned int)((gaddr >> 32) & 0x1FFFFFFull) | (2u << 30);  // type=2
  i32x8 g1;
  g1[0] = (int)((2u << 16) |                    // data_size = 4 bytes
                (pad_enable << 20) |
                (pad_interval_code << 22) |
                (pad_amount_code << 25));
  g1[1] = (int)(tensor_d0 << 16);               // tensor_dim0[15:0] at [63:48]
  g1[2] = (int)((tensor_d0 >> 16) | (tensor_d1 << 16));
  g1[3] = (int)((tensor_d1 >> 16) | (tile_d0 << 16));
  g1[4] = (int)tile_d1;                         // tile_dim1 (tile_dim2 = 0)
  g1[5] = (int)stride0;                         // tensor_dim0_stride[31:0]
  g1[6] = 0;                                    // stride0 hi | stride1 lo
  g1[7] = 0;
  asm volatile("tensor_load_to_lds %0, %1" ::"s"(g0), "s"(g1) : "memory");
}

// ---------------------------------------------------------------------------
// Kernel 1: GroupNorm (32 groups of 8 channels) -> h (f16, [B][C][N])
// ---------------------------------------------------------------------------
__global__ void gn_kernel(const float* __restrict__ x,
                          const float* __restrict__ gnw,
                          const float* __restrict__ gnb,
                          _Float16* __restrict__ h) {
  const int b = blockIdx.x >> 5;
  const int g = blockIdx.x & 31;
  const int cg = g * 8;
  const float* xg = x + ((size_t)b * C_DIM + cg) * N_DIM;
  _Float16* hg = h + ((size_t)b * C_DIM + cg) * N_DIM;
  const int tid = threadIdx.x;
  const int lane = tid & 31, w = tid >> 5;

  float s = 0.f, s2 = 0.f;
  for (int i = tid; i < 8 * N_DIM; i += 256) {
    float v = xg[i];
    s += v;
    s2 += v * v;
  }
#pragma unroll
  for (int off = 16; off > 0; off >>= 1) {
    s += __shfl_xor(s, off);
    s2 += __shfl_xor(s2, off);
  }
  __shared__ float rS[8], rS2[8];
  if (lane == 0) { rS[w] = s; rS2[w] = s2; }
  __syncthreads();
  if (tid == 0) {
    float S = 0.f, S2 = 0.f;
#pragma unroll
    for (int i = 0; i < 8; ++i) { S += rS[i]; S2 += rS2[i]; }
    rS[0] = S;
    rS2[0] = S2;
  }
  __syncthreads();
  const float inv_n = 1.0f / (8.0f * N_DIM);
  const float mean = rS[0] * inv_n;
  const float var = rS2[0] * inv_n - mean * mean;
  const float rstd = rsqrtf(var + 1e-5f);
  for (int i = tid; i < 8 * N_DIM; i += 256) {
    int c = cg + (i >> 12);
    hg[i] = (_Float16)((xg[i] - mean) * rstd * gnw[c] + gnb[c]);
  }
}

// ---------------------------------------------------------------------------
// Kernel 2: fused Q/K/V projection.  QT[n][d] = sum_c h[c][n] * W[c][d] + b[d]
// Block: 64 n-rows x 64 d-cols, 8 waves (4 n-groups x 2 d-halves), K-loop of 8.
// Stores qT,kT as [B][N][C] f16 and v as [B][C][N] f16.
// ---------------------------------------------------------------------------
__global__ void proj_qkv_kernel(const _Float16* __restrict__ h,
                                const float* __restrict__ Wq, const float* __restrict__ bq,
                                const float* __restrict__ Wk, const float* __restrict__ bk,
                                const float* __restrict__ Wv, const float* __restrict__ bv,
                                _Float16* __restrict__ qT,
                                _Float16* __restrict__ kT,
                                _Float16* __restrict__ vC) {
  const int b = blockIdx.z;
  const int n0 = blockIdx.x * 64;
  const int d0 = blockIdx.y * 64;
  const int tid = threadIdx.x;
  const int lane = tid & 31, w = tid >> 5;
  const int ngrp = w & 3, dhalf = w >> 2;
  const int lm = lane & 15, hi = lane >> 4;

  __shared__ _Float16 sH[32][72];        // h tile [c_local][n_local]
  __shared__ _Float16 sWt[3][64][40];    // W tiles transposed [mat][d_local][c_local]

  const _Float16* hB = h + (size_t)b * C_DIM * N_DIM;
  const float* Wm[3] = {Wq, Wk, Wv};

  v8f acc[3][2] = {};

  for (int kb = 0; kb < 8; ++kb) {
    const int c0 = kb * 32;
    {  // stage h tile: 32 c x 64 n, coalesced 16B per thread
      int r = tid >> 3;
      int cc = (tid & 7) * 8;
      *(float4*)&sH[r][cc] =
          *(const float4*)(hB + (size_t)(c0 + r) * N_DIM + n0 + cc);
      if (kb < 7)  // prefetch next h tile into cache hierarchy
        __builtin_prefetch(hB + (size_t)(c0 + 32 + r) * N_DIM + n0 + cc, 0, 3);
    }
#pragma unroll
    for (int m = 0; m < 3; ++m) {  // stage transposed f16 weight tiles
      int r = tid >> 3;            // c_local
      int dseg = (tid & 7) * 8;
      const float* src = Wm[m] + (size_t)(c0 + r) * C_DIM + d0 + dseg;
      float4 v0 = ((const float4*)src)[0];
      float4 v1 = ((const float4*)src)[1];
      sWt[m][dseg + 0][r] = (_Float16)v0.x;
      sWt[m][dseg + 1][r] = (_Float16)v0.y;
      sWt[m][dseg + 2][r] = (_Float16)v0.z;
      sWt[m][dseg + 3][r] = (_Float16)v0.w;
      sWt[m][dseg + 4][r] = (_Float16)v1.x;
      sWt[m][dseg + 5][r] = (_Float16)v1.y;
      sWt[m][dseg + 6][r] = (_Float16)v1.z;
      sWt[m][dseg + 7][r] = (_Float16)v1.w;
    }
    __syncthreads();

    // A fragment: 16 n-rows of H^T, K = 32 channels (A-layout gather)
    v16h a;
#pragma unroll
    for (int e = 0; e < 16; ++e) {
      int k = (hi << 4) + e;
      a[e] = sH[k][ngrp * 16 + lm];
    }
#pragma unroll
    for (int m = 0; m < 3; ++m) {
#pragma unroll
      for (int t = 0; t < 2; ++t) {
        int drow = dhalf * 32 + t * 16 + lm;
        const _Float16* rp = &sWt[m][drow][hi * 16];
        v16h bb = pack16(*(const v8h*)rp, *(const v8h*)(rp + 8));
        acc[m][t] = wmma_f16(a, bb, acc[m][t]);
      }
    }
    __syncthreads();
  }

#pragma unroll
  for (int t = 0; t < 2; ++t) {
    const int dcol = d0 + dhalf * 32 + t * 16 + lm;
    const float biasq = bq[dcol], biask = bk[dcol], biasv = bv[dcol];
#pragma unroll
    for (int j = 0; j < 8; ++j) {
      const int n = n0 + ngrp * 16 + j + hi * 8;
      qT[((size_t)b * N_DIM + n) * C_DIM + dcol] = (_Float16)(acc[0][t][j] + biasq);
      kT[((size_t)b * N_DIM + n) * C_DIM + dcol] = (_Float16)(acc[1][t][j] + biask);
      vC[((size_t)b * C_DIM + dcol) * N_DIM + n] = (_Float16)(acc[2][t][j] + biasv);
    }
  }
}

// ---------------------------------------------------------------------------
// Kernel 3: flash attention.  Block = 64 query rows, streams 32 keys / iter.
// 8 waves = 4 query groups (16 rows) x 2 channel halves (128 c each).
// K/V tiles are DMA'd into LDS by the Tensor Data Mover (padded rows),
// never materializing the 4096x4096 logits (saves ~800MB of HBM traffic).
// ---------------------------------------------------------------------------
__global__ void attn_kernel(const _Float16* __restrict__ qT,
                            const _Float16* __restrict__ kT,
                            const _Float16* __restrict__ vC,
                            _Float16* __restrict__ oT) {
  const int b = blockIdx.y;
  const int n0 = blockIdx.x * 64;
  const int tid = threadIdx.x;
  const int lane = tid & 31, w = tid >> 5;
  const int ngrp = w & 3, chalf = w >> 2;
  const int lm = lane & 15, hi = lane >> 4;

  __shared__ _Float16 sK[32][264];     // key tile  [key][c], +8 halves pad
  __shared__ _Float16 sVc[256][40];    // value tile [c][key], +8 halves pad
  __shared__ _Float16 sP[8][16][40];   // per-wave P exchange (D->A relayout)

  // Hoist Q fragments: this wave's 16 query rows x K=256 -> 8 A fragments
  v16h qa[8];
  {
    const _Float16* qrow =
        qT + ((size_t)b * N_DIM + n0 + ngrp * 16 + lm) * C_DIM;
#pragma unroll
    for (int kb = 0; kb < 8; ++kb)
      qa[kb] = pack16(*(const v8h*)(qrow + kb * 32 + hi * 16),
                      *(const v8h*)(qrow + kb * 32 + hi * 16 + 8));
  }

  const unsigned int ldsK = (unsigned int)(unsigned long long)&sK[0][0];
  const unsigned int ldsV = (unsigned int)(unsigned long long)&sVc[0][0];
  const unsigned long long gK =
      (unsigned long long)(const void*)(kT + (size_t)b * N_DIM * C_DIM);
  const unsigned long long gV =
      (unsigned long long)(const void*)(vC + (size_t)b * C_DIM * N_DIM);

  v8f o[8] = {};
  float rmax[8], rsum[8];
#pragma unroll
  for (int j = 0; j < 8; ++j) { rmax[j] = -1e30f; rsum[j] = 0.f; }

  const float scale = 0.0625f;  // C^-0.5

  for (int m0 = 0; m0 < N_DIM; m0 += 32) {
    __syncthreads();  // previous iteration done with sK/sVc/sP
    if (w == 0) {
      // K tile: 32 key rows x 128 dwords; LDS rows padded 128+4 dwords
      // (pad_interval code 3->? : interval dwords = 2<<code ; 2<<6 = 128)
      tdm_load_2d(gK + (size_t)m0 * (C_DIM * 2), ldsK,
                  /*tensor_d0=*/C_DIM / 2, /*tensor_d1=*/N_DIM,
                  /*stride0=*/C_DIM / 2,
                  /*tile_d0=*/C_DIM / 2, /*tile_d1=*/32,
                  /*pad_interval=*/6, /*pad_amount=*/3, /*pad_enable=*/1);
      // V tile: 256 c rows x 16 dwords; LDS rows padded 16+4 dwords (2<<3=16)
      tdm_load_2d(gV + (size_t)m0 * 2, ldsV,
                  /*tensor_d0=*/N_DIM / 2, /*tensor_d1=*/C_DIM,
                  /*stride0=*/N_DIM / 2,
                  /*tile_d0=*/16, /*tile_d1=*/C_DIM,
                  /*pad_interval=*/3, /*pad_amount=*/3, /*pad_enable=*/1);
      __builtin_amdgcn_s_wait_tensorcnt(0);
    }
    __syncthreads();

    // S = Q^T K : 16 query rows x 32 keys (two 16x16 tiles), K-dim = 256
    v8f s0 = {}, s1 = {};
#pragma unroll
    for (int kb = 0; kb < 8; ++kb) {
      const _Float16* k0 = &sK[lm][kb * 32 + hi * 16];
      const _Float16* k1 = &sK[16 + lm][kb * 32 + hi * 16];
      v16h b0 = pack16(*(const v8h*)k0, *(const v8h*)(k0 + 8));
      v16h b1 = pack16(*(const v8h*)k1, *(const v8h*)(k1 + 8));
      s0 = wmma_f16(qa[kb], b0, s0);
      s1 = wmma_f16(qa[kb], b1, s1);
    }

    // online softmax (row = query): max/sum reductions across the 16-lane half
    float alpha[8];
#pragma unroll
    for (int j = 0; j < 8; ++j) {
      float a0 = s0[j] * scale;
      float a1 = s1[j] * scale;
      float mx = fmaxf(a0, a1);
      mx = fmaxf(mx, __shfl_xor(mx, 1));
      mx = fmaxf(mx, __shfl_xor(mx, 2));
      mx = fmaxf(mx, __shfl_xor(mx, 4));
      mx = fmaxf(mx, __shfl_xor(mx, 8));
      float nm = fmaxf(rmax[j], mx);
      float al = __expf(rmax[j] - nm);
      float p0 = __expf(a0 - nm);
      float p1 = __expf(a1 - nm);
      float ps = p0 + p1;
      ps += __shfl_xor(ps, 1);
      ps += __shfl_xor(ps, 2);
      ps += __shfl_xor(ps, 4);
      ps += __shfl_xor(ps, 8);
      rmax[j] = nm;
      rsum[j] = rsum[j] * al + ps;
      alpha[j] = al;
      sP[w][j + hi * 8][lm] = (_Float16)p0;       // D-layout -> logical P
      sP[w][j + hi * 8][16 + lm] = (_Float16)p1;
    }
#pragma unroll
    for (int ct = 0; ct < 8; ++ct)
#pragma unroll
      for (int j = 0; j < 8; ++j) o[ct][j] *= alpha[j];

    __syncthreads();  // make P relayout visible (and ordered) before A reads

    // P fragment in A-layout (16 rows x 32 keys)
    const _Float16* pr = &sP[w][lm][0];
    v16h pa = pack16(*(const v8h*)(pr + hi * 16), *(const v8h*)(pr + hi * 16 + 8));

    // O += P * V for this wave's 128 channels (8 tiles)
#pragma unroll
    for (int ct = 0; ct < 8; ++ct) {
      const int ccol = chalf * 128 + ct * 16 + lm;
      const _Float16* vr = &sVc[ccol][hi * 16];
      v16h bv = pack16(*(const v8h*)vr, *(const v8h*)(vr + 8));
      o[ct] = wmma_f16(pa, bv, o[ct]);
    }
  }

  // normalize and store O^T as [B][N][C] f16
#pragma unroll
  for (int ct = 0; ct < 8; ++ct) {
    const int ccol = chalf * 128 + ct * 16 + lm;
#pragma unroll
    for (int j = 0; j < 8; ++j) {
      const int n = n0 + ngrp * 16 + j + hi * 8;
      oT[((size_t)b * N_DIM + n) * C_DIM + ccol] = (_Float16)(o[ct][j] / rsum[j]);
    }
  }
}

// ---------------------------------------------------------------------------
// Kernel 4: out = (x + O^T * Wp + bp) / sqrt(2)
// ---------------------------------------------------------------------------
__global__ void proj_out_kernel(const _Float16* __restrict__ oT,
                                const float* __restrict__ Wp,
                                const float* __restrict__ bp,
                                const float* __restrict__ x,
                                float* __restrict__ out) {
  const int b = blockIdx.z;
  const int n0 = blockIdx.x * 64;
  const int d0 = blockIdx.y * 64;
  const int tid = threadIdx.x;
  const int lane = tid & 31, w = tid >> 5;
  const int ngrp = w & 3, dhalf = w >> 2;
  const int lm = lane & 15, hi = lane >> 4;

  __shared__ _Float16 sWt[64][264];  // Wp slice transposed [d_local][c]

  {  // stage entire 256 x 64 weight slice once (f32 -> f16)
    const float* wr = Wp + (size_t)tid * C_DIM + d0;
#pragma unroll
    for (int i = 0; i < 16; ++i) {
      float4 v = ((const float4*)wr)[i];
      sWt[i * 4 + 0][tid] = (_Float16)v.x;
      sWt[i * 4 + 1][tid] = (_Float16)v.y;
      sWt[i * 4 + 2][tid] = (_Float16)v.z;
      sWt[i * 4 + 3][tid] = (_Float16)v.w;
    }
  }

  // hoist A fragments (oT rows are already [n][c])
  v16h af[8];
  {
    const _Float16* arow =
        oT + ((size_t)b * N_DIM + n0 + ngrp * 16 + lm) * C_DIM;
#pragma unroll
    for (int kb = 0; kb < 8; ++kb)
      af[kb] = pack16(*(const v8h*)(arow + kb * 32 + hi * 16),
                      *(const v8h*)(arow + kb * 32 + hi * 16 + 8));
  }
  __syncthreads();

  v8f acc[2] = {};
#pragma unroll
  for (int kb = 0; kb < 8; ++kb) {
#pragma unroll
    for (int t = 0; t < 2; ++t) {
      const int drow = dhalf * 32 + t * 16 + lm;
      const _Float16* rp = &sWt[drow][kb * 32 + hi * 16];
      v16h bb = pack16(*(const v8h*)rp, *(const v8h*)(rp + 8));
      acc[t] = wmma_f16(af[kb], bb, acc[t]);
    }
  }

  const float inv_sqrt2 = 0.70710678118654752f;
#pragma unroll
  for (int t = 0; t < 2; ++t) {
    const int d = d0 + dhalf * 32 + t * 16 + lm;
    const float bias = bp[d];
#pragma unroll
    for (int j = 0; j < 8; ++j) {
      const int n = n0 + ngrp * 16 + j + hi * 8;
      const size_t idx = ((size_t)b * C_DIM + d) * N_DIM + n;
      out[idx] = (x[idx] + acc[t][j] + bias) * inv_sqrt2;
    }
  }
}

// ---------------------------------------------------------------------------
extern "C" void kernel_launch(void* const* d_in, const int* in_sizes, int n_in,
                              void* d_out, int out_size, void* d_ws, size_t ws_size,
                              hipStream_t stream) {
  const float* x    = (const float*)d_in[0];
  const float* gn_w = (const float*)d_in[1];
  const float* gn_b = (const float*)d_in[2];
  const float* Wq   = (const float*)d_in[3];
  const float* bq   = (const float*)d_in[4];
  const float* Wk   = (const float*)d_in[5];
  const float* bk   = (const float*)d_in[6];
  const float* Wv   = (const float*)d_in[7];
  const float* bv   = (const float*)d_in[8];
  const float* Wp   = (const float*)d_in[9];
  const float* bp   = (const float*)d_in[10];
  float* out = (float*)d_out;

  const size_t elems = (size_t)B_DIM * C_DIM * N_DIM;  // 4M elements
  _Float16* h  = (_Float16*)d_ws;                      // 8 MB each
  _Float16* qT = h + elems;
  _Float16* kT = qT + elems;
  _Float16* vC = kT + elems;
  _Float16* oT = vC + elems;                           // total 40 MB workspace

  gn_kernel<<<dim3(B_DIM * 32), 256, 0, stream>>>(x, gn_w, gn_b, h);
  proj_qkv_kernel<<<dim3(64, 4, B_DIM), 256, 0, stream>>>(
      h, Wq, bq, Wk, bk, Wv, bv, qT, kT, vC);
  attn_kernel<<<dim3(64, B_DIM), 256, 0, stream>>>(qT, kT, vC, oT);
  proj_out_kernel<<<dim3(64, 4, B_DIM), 256, 0, stream>>>(oT, Wp, bp, x, out);
}
